// Word2Vec_88639535054896
// MI455X (gfx1250) — matrix operand order repaired and checked by reading, not compiled, not run on hardware
//
#include <hip/hip_runtime.h>
#include <hip/hip_bf16.h>
#include <math.h>

// CDNA5 / gfx1250 wave32 WMMA types
typedef __attribute__((ext_vector_type(16))) __bf16 v16bf;
typedef __attribute__((ext_vector_type(8)))  float  v8f;
typedef __attribute__((ext_vector_type(4)))  float  v4f;

#define DIM   256
#define NEGK  5

__device__ __forceinline__ float log_sigmoid(float x) {
    // log(sigmoid(x)) = min(x,0) - log1p(exp(-|x|))
    return fminf(x, 0.0f) - log1pf(__expf(-fabsf(x)));
}

__global__ void zero_out_kernel(float* out) {
    if (threadIdx.x == 0 && blockIdx.x == 0) out[0] = 0.0f;
}

// One wave handles 16 (target, context, 5-neg) pairs per group iteration.
// B operand  = 16 target embeddings (emb_table rows) as columns, kept in regs
//              for all 8 K-chunks of 32.
// A operand  = 16 slot embeddings (out_table rows: ctx or neg_k) as rows.
// Score for pair i = diagonal element C[i][i] of the 16x16 WMMA result.
__global__ void __launch_bounds__(256)
w2v_sgns_wmma(const int* __restrict__ tgt_ids,
              const int* __restrict__ ctx_ids,
              const int* __restrict__ neg_ids,
              const float* __restrict__ emb_table,
              const float* __restrict__ out_table,
              float* __restrict__ out,
              int N) {
    const int lane = threadIdx.x & 31;
    const int m    = lane & 15;   // pair slot within group (row/col index)
    const int h    = lane >> 4;   // half selector for K-splitting
    const int wavesPerBlock = blockDim.x >> 5;
    const int gwave  = blockIdx.x * wavesPerBlock + (threadIdx.x >> 5);
    const int nwaves = gridDim.x * wavesPerBlock;
    const int ngroups = (N + 15) >> 4;

    float acc = 0.0f;

    for (int g = gwave; g < ngroups; g += nwaves) {
        const int  p     = g * 16 + m;
        const bool valid = (p < N) && (h == 0);
        const int  pc    = (p < N) ? p : (N - 1);

        // ---- Preload B fragments: target embedding of pair m ----
        // B layout (32x16 bf16): lane n<16 holds col n, K = 0..15 (VGPR0..7,
        // 2 packed per VGPR); lane n+16 holds col n, K = 16..31.
        const int tid = tgt_ids[pc];
        const float* trow = emb_table + (size_t)tid * DIM;
        v16bf bfrag[8];
        #pragma unroll
        for (int c = 0; c < 8; ++c) {
            const float* pB = trow + c * 32 + h * 16;   // 16 contiguous floats
            v4f x0 = *(const v4f*)(pB + 0);
            v4f x1 = *(const v4f*)(pB + 4);
            v4f x2 = *(const v4f*)(pB + 8);
            v4f x3 = *(const v4f*)(pB + 12);
            #pragma unroll
            for (int j = 0; j < 4; ++j) {
                bfrag[c][j]      = (__bf16)x0[j];
                bfrag[c][4 + j]  = (__bf16)x1[j];
                bfrag[c][8 + j]  = (__bf16)x2[j];
                bfrag[c][12 + j] = (__bf16)x3[j];
            }
        }

        float lsum = 0.0f;

        // slot 0 = positive context, slots 1..5 = negatives
        #pragma unroll
        for (int s = 0; s < 6; ++s) {
            const int sid = (s == 0) ? ctx_ids[pc]
                                     : neg_ids[(size_t)pc * NEGK + (s - 1)];
            const float* srow = out_table + (size_t)sid * DIM;

            v8f cacc = {};   // zero accumulator
            #pragma unroll
            for (int c = 0; c < 8; ++c) {
                // A layout (16x32 bf16): lane (m,h) holds row m,
                // K = {c*32 + h*16 .. +15} (contiguous, matching B).
                const float* pA0 = srow + c * 32 + h * 16;
                const float* pA1 = srow + c * 32 + h * 16 + 8;
                v4f a0 = *(const v4f*)(pA0 + 0);
                v4f a1 = *(const v4f*)(pA0 + 4);
                v4f a2 = *(const v4f*)(pA1 + 0);
                v4f a3 = *(const v4f*)(pA1 + 4);
                v16bf afrag;
                #pragma unroll
                for (int j = 0; j < 4; ++j) {
                    afrag[j]      = (__bf16)a0[j];
                    afrag[4 + j]  = (__bf16)a1[j];
                    afrag[8 + j]  = (__bf16)a2[j];
                    afrag[12 + j] = (__bf16)a3[j];
                }
                // D = A x B + C  (v_wmma_f32_16x16x32_bf16)
                cacc = __builtin_amdgcn_wmma_f32_16x16x32_bf16(
                    /*neg_a=*/false, afrag, /*neg_b=*/false, bfrag[c],
                    /*c_mod=*/(short)0, cacc,
                    /*reuse_a=*/false, /*reuse_b=*/false);
            }

            // Diagonal extraction: C[i][i] lives at VGPR r = i&7,
            // lane = (i<8 ? i : i+16). Lane j (0..15) pulls its own score.
            const int src = (lane < 8) ? lane : lane + 16;
            float diag = 0.0f;
            #pragma unroll
            for (int r = 0; r < 8; ++r) {
                float t = __shfl(cacc[r], src, 32);
                if ((lane & 7) == r) diag = t;
            }

            const float x = (s == 0) ? diag : -diag;
            lsum += log_sigmoid(x);
        }

        if (valid) acc += -lsum;   // loss_per_pair = -(ls_pos + sum ls_neg)
    }

    // Wave-level reduction across 32 lanes (lanes 16..31 hold 0)
    #pragma unroll
    for (int off = 16; off >= 1; off >>= 1)
        acc += __shfl_down(acc, off, 32);

    if (lane == 0)
        atomicAdd(out, acc / (float)N);   // mean over pairs
}

extern "C" void kernel_launch(void* const* d_in, const int* in_sizes, int n_in,
                              void* d_out, int out_size, void* d_ws, size_t ws_size,
                              hipStream_t stream) {
    const int*   tgt_ids   = (const int*)d_in[0];
    const int*   ctx_ids   = (const int*)d_in[1];
    const int*   neg_ids   = (const int*)d_in[2];
    const float* emb_table = (const float*)d_in[3];
    const float* out_table = (const float*)d_in[4];
    float*       out       = (float*)d_out;
    const int    N         = in_sizes[0];

    zero_out_kernel<<<1, 64, 0, stream>>>(out);

    // 512 blocks x 8 waves = 4096 waves; 12500 groups of 16 pairs -> ~3 each
    const int threads = 256;
    const int blocks  = 512;
    w2v_sgns_wmma<<<blocks, threads, 0, stream>>>(
        tgt_ids, ctx_ids, neg_ids, emb_table, out_table, out, N);
}
